// AutoCorrelation_6451040879023
// MI455X (gfx1250) — matrix operand order repaired and checked by reading, not compiled, not run on hardware
//
#include <hip/hip_runtime.h>
#include <hip/hip_bf16.h>

// ---------------------------------------------------------------------------
// AutoCorrelation (Autoformer) for MI455X / gfx1250, wave32, WMMA bf16.
//   1) k_cvt_bf16  : q,k f32 -> bf16 (memory-bound)
//   2) k_corr_bf16 : wrapped-diagonal sums of C = Q K^T via
//                    V_WMMA_F32_16X16X32_BF16. Each wave owns 4 consecutive
//                    block-diagonals + a 16-tile ti strip; a 4-slot rolling
//                    register window of B fragments gives 4 WMMAs per
//                    (1 A-load + 1 B-load)  ->  16 B/lane/WMMA.
//   3) k_topk      : combine partials, batch-mean, top-7, per-batch softmax
//   4) k_agg       : weighted circular gather of values
// ---------------------------------------------------------------------------

typedef __bf16 bf16_t;
typedef __attribute__((ext_vector_type(16))) __bf16 v16bf;
typedef __attribute__((ext_vector_type(8)))  __bf16 v8bf;
typedef __attribute__((ext_vector_type(8)))  float  v8f;

constexpr int SEQ = 2048;   // L
constexpr int CHN = 512;    // H*E (contiguous innermost in (B,L,H,E))
constexpr int NB  = 8;      // batches
constexpr int NT  = 128;    // SEQ/16 tiles along L
constexpr int TK  = 7;      // int(log(2048)) top-k lags

// -------------------------- stage 1: f32 -> bf16 ---------------------------
__global__ __launch_bounds__(256)
void k_cvt_bf16(const float* __restrict__ src, bf16_t* __restrict__ dst, int n8) {
  int i = blockIdx.x * 256 + threadIdx.x;
  if (i >= n8) return;
  const float4* s = (const float4*)src + (size_t)i * 2;
  float4 a = s[0], b = s[1];
  v8bf o;
  o[0] = (bf16_t)a.x; o[1] = (bf16_t)a.y; o[2] = (bf16_t)a.z; o[3] = (bf16_t)a.w;
  o[4] = (bf16_t)b.x; o[5] = (bf16_t)b.y; o[6] = (bf16_t)b.z; o[7] = (bf16_t)b.w;
  *(v8bf*)(dst + (size_t)i * 8) = o;
}

// ----------------- WMMA fragment loaders (wave32, contiguous k = 16h+s) ---------------
// A (16x32 MxK): lane m=lane&15 -> row M=m; K = kb..kb+15,
//                kb = 16*(lane>=16), packed low-to-high.
// B (32x16 KxN): lane -> col N=lane&15; K = kb2..kb2+15, kb2 = 16*(lane>=16).
__device__ __forceinline__ v16bf loadA_bf(const bf16_t* __restrict__ Qb,
                                          int tile, int m, int abk, int e0) {
  const bf16_t* p = Qb + (size_t)(16 * tile + m) * CHN + e0 + abk;
  v8bf a0 = *(const v8bf*)p;
  v8bf a1 = *(const v8bf*)(p + 8);
  v16bf A;
  #pragma unroll
  for (int i = 0; i < 8; ++i) { A[i] = a0[i]; A[8 + i] = a1[i]; }
  return A;
}
__device__ __forceinline__ v16bf loadB_bf(const bf16_t* __restrict__ Kb,
                                          int tile, int n, int bbk, int e0) {
  tile &= NT - 1;   // circular wrap (works for negative tile too)
  return *(const v16bf*)(Kb + (size_t)(16 * tile + n) * CHN + e0 + bbk);
}

// -------------------------- stage 2: WMMA correlation ----------------------
// Block (dg, b): diagonals d0..d0+3, d0 = 4*dg. Wave w owns ti in [16w,16w+16).
// For pair (ti, tj=(ti-d)&127): element (M,N) -> tau = 16d + M - N, independent
// of ti, so one accumulator per diagonal survives the whole strip.
__global__ __launch_bounds__(256)
void k_corr_bf16(const bf16_t* __restrict__ Q, const bf16_t* __restrict__ K,
                 float* __restrict__ P) {
  const int d0   = blockIdx.x * 4;      // first diagonal of this WG
  const int b    = blockIdx.y;
  const int lane = threadIdx.x & 31;
  const int wave = threadIdx.x >> 5;    // 8 waves
  const int t0   = wave << 4;           // ti strip base

  const int m   = lane & 15;            // A row within tile == B col within tile
  const int hi  = lane >> 4;
  const int abk = hi << 4;              // 0 or 16
  const int bbk = hi << 4;              // 0 or 16

  const bf16_t* Qb = Q + (size_t)b * SEQ * CHN;
  const bf16_t* Kb = K + (size_t)b * SEQ * CHN;

  v8f acc0 = {}, acc1 = {}, acc2 = {}, acc3 = {};

#define CORR_STEP(BNEW, BJ1, BJ2, BJ3, II)                                     \
  do {                                                                         \
    BNEW = loadB_bf(Kb, t0 + (II) - d0, m, bbk, e0);                           \
    v16bf Afr = loadA_bf(Qb, t0 + (II), m, abk, e0);                           \
    acc0 = __builtin_amdgcn_wmma_f32_16x16x32_bf16(false, Afr, false, BNEW,    \
                                                   (short)0, acc0, false, false); \
    acc1 = __builtin_amdgcn_wmma_f32_16x16x32_bf16(false, Afr, false, BJ1,     \
                                                   (short)0, acc1, false, false); \
    acc2 = __builtin_amdgcn_wmma_f32_16x16x32_bf16(false, Afr, false, BJ2,     \
                                                   (short)0, acc2, false, false); \
    acc3 = __builtin_amdgcn_wmma_f32_16x16x32_bf16(false, Afr, false, BJ3,     \
                                                   (short)0, acc3, false, false); \
  } while (0)

  for (int e0 = 0; e0 < CHN; e0 += 32) {
    v16bf Bw0, Bw1, Bw2, Bw3;
    // Prologue: window holds tiles t0-d0-3 .. t0-d0-1 in slots 1,2,3.
    Bw1 = loadB_bf(Kb, t0 - d0 - 3, m, bbk, e0);
    Bw2 = loadB_bf(Kb, t0 - d0 - 2, m, bbk, e0);
    Bw3 = loadB_bf(Kb, t0 - d0 - 1, m, bbk, e0);
    #pragma unroll 1
    for (int i = 0; i < 16; i += 4) {
      CORR_STEP(Bw0, Bw3, Bw2, Bw1, i);
      CORR_STEP(Bw1, Bw0, Bw3, Bw2, i + 1);
      CORR_STEP(Bw2, Bw1, Bw0, Bw3, i + 2);
      CORR_STEP(Bw3, Bw2, Bw1, Bw0, i + 3);
    }
  }
#undef CORR_STEP

  // Diagonal reduction. C/D layout: reg r, lane -> M = r + 8*(lane>=16),
  // N = lane&15; local delta = M - N in [-15,15].
  __shared__ float diag[4][31];
  const int tid = threadIdx.x;
  if (tid < 124) diag[tid / 31][tid % 31] = 0.0f;
  __syncthreads();
  const int Mb = hi << 3;
  #pragma unroll
  for (int r = 0; r < 8; ++r) {
    const int dl = Mb + r - m + 15;
    atomicAdd(&diag[0][dl], acc0[r]);
    atomicAdd(&diag[1][dl], acc1[r]);
    atomicAdd(&diag[2][dl], acc2[r]);
    atomicAdd(&diag[3][dl], acc3[r]);
  }
  __syncthreads();
  if (tid < 124) {
    P[((size_t)b * NT + d0 + tid / 31) * 31 + tid % 31] =
        diag[tid / 31][tid % 31] * (1.0f / 512.0f);
  }
}

// ------------- f32-source fallback (used only if ws is too small) ----------
__global__ __launch_bounds__(128)
void k_corr_f32(const float* __restrict__ Q, const float* __restrict__ K,
                float* __restrict__ P) {
  const int d    = blockIdx.x;
  const int b    = blockIdx.y;
  const int lane = threadIdx.x & 31;
  const int wave = threadIdx.x >> 5;

  const int m   = lane & 15;
  const int hi  = lane >> 4;
  const int abk = hi << 4;
  const int bbk = hi << 4;

  const float* Qb = Q + (size_t)b * SEQ * CHN;
  const float* Kb = K + (size_t)b * SEQ * CHN;

  v8f acc = {};
  for (int t = 0; t < 32; ++t) {
    const int ti = wave + 4 * t;
    const int tj = (ti - d) & (NT - 1);
    const float* qrow = Qb + (size_t)(16 * ti + m) * CHN;
    const float* krow = Kb + (size_t)(16 * tj + m) * CHN;
    #pragma unroll 2
    for (int e0 = 0; e0 < CHN; e0 += 32) {
      float4 f0 = *(const float4*)(qrow + e0 + abk);
      float4 f1 = *(const float4*)(qrow + e0 + abk + 4);
      float4 f2 = *(const float4*)(qrow + e0 + abk + 8);
      float4 f3 = *(const float4*)(qrow + e0 + abk + 12);
      v16bf A;
      A[0]=(bf16_t)f0.x; A[1]=(bf16_t)f0.y; A[2]=(bf16_t)f0.z; A[3]=(bf16_t)f0.w;
      A[4]=(bf16_t)f1.x; A[5]=(bf16_t)f1.y; A[6]=(bf16_t)f1.z; A[7]=(bf16_t)f1.w;
      A[8]=(bf16_t)f2.x; A[9]=(bf16_t)f2.y; A[10]=(bf16_t)f2.z; A[11]=(bf16_t)f2.w;
      A[12]=(bf16_t)f3.x; A[13]=(bf16_t)f3.y; A[14]=(bf16_t)f3.z; A[15]=(bf16_t)f3.w;
      float4 g0 = *(const float4*)(krow + e0 + bbk);
      float4 g1 = *(const float4*)(krow + e0 + bbk + 4);
      float4 g2 = *(const float4*)(krow + e0 + bbk + 8);
      float4 g3 = *(const float4*)(krow + e0 + bbk + 12);
      v16bf B;
      B[0]=(bf16_t)g0.x; B[1]=(bf16_t)g0.y; B[2]=(bf16_t)g0.z; B[3]=(bf16_t)g0.w;
      B[4]=(bf16_t)g1.x; B[5]=(bf16_t)g1.y; B[6]=(bf16_t)g1.z; B[7]=(bf16_t)g1.w;
      B[8]=(bf16_t)g2.x; B[9]=(bf16_t)g2.y; B[10]=(bf16_t)g2.z; B[11]=(bf16_t)g2.w;
      B[12]=(bf16_t)g3.x; B[13]=(bf16_t)g3.y; B[14]=(bf16_t)g3.z; B[15]=(bf16_t)g3.w;
      acc = __builtin_amdgcn_wmma_f32_16x16x32_bf16(
                false, A, false, B, (short)0, acc, false, false);
    }
  }

  __shared__ float diag[4][31];
  if (threadIdx.x < 124) diag[threadIdx.x / 31][threadIdx.x % 31] = 0.0f;
  __syncthreads();
  const int Mb = hi << 3;
  #pragma unroll
  for (int r = 0; r < 8; ++r) {
    atomicAdd(&diag[wave][Mb + r - m + 15], acc[r]);
  }
  __syncthreads();
  if (threadIdx.x < 31) {
    float s = diag[0][threadIdx.x] + diag[1][threadIdx.x]
            + diag[2][threadIdx.x] + diag[3][threadIdx.x];
    P[((size_t)b * NT + d) * 31 + threadIdx.x] = s * (1.0f / 512.0f);
  }
}

// mean_value[b][tau] from the two overlapping block-diagonal partials
__device__ inline float mv_from_P(const float* __restrict__ P, int b, int tau) {
  int dd = tau >> 4, r = tau & 15;
  float v = P[((size_t)b * NT + dd) * 31 + (r + 15)];
  if (r >= 1) v += P[((size_t)b * NT + ((dd + 1) & (NT - 1))) * 31 + (r - 1)];
  return v;
}

// -------------------------- stage 3: top-k + softmax -----------------------
__global__ __launch_bounds__(256)
void k_topk(const float* __restrict__ P, int* __restrict__ idx_out,
            float* __restrict__ w_out) {
  __shared__ float sc[SEQ];
  __shared__ float rv[256];
  __shared__ int   ri[256];
  __shared__ int   chosen[TK];
  const int tid = threadIdx.x;

  for (int t = tid; t < SEQ; t += 256) {
    float s = 0.0f;
    for (int b = 0; b < NB; ++b) s += mv_from_P(P, b, t);
    sc[t] = s * (1.0f / NB);
  }
  __syncthreads();

  for (int it = 0; it < TK; ++it) {
    float best = -3.4e38f; int bi = 0;
    for (int t = tid; t < SEQ; t += 256) {
      float v = sc[t];
      if (v > best) { best = v; bi = t; }
    }
    rv[tid] = best; ri[tid] = bi;
    __syncthreads();
    for (int s = 128; s > 0; s >>= 1) {
      if (tid < s) {
        float v = rv[tid + s]; int j = ri[tid + s];
        if (v > rv[tid] || (v == rv[tid] && j < ri[tid])) { rv[tid] = v; ri[tid] = j; }
      }
      __syncthreads();
    }
    if (tid == 0) { chosen[it] = ri[0]; sc[ri[0]] = -3.4e38f; }
    __syncthreads();
  }

  if (tid < TK) idx_out[tid] = chosen[tid];
  if (tid < NB) {
    float v[TK]; float mx = -3.4e38f;
    #pragma unroll
    for (int k2 = 0; k2 < TK; ++k2) {
      v[k2] = mv_from_P(P, tid, chosen[k2]);
      if (v[k2] > mx) mx = v[k2];
    }
    float s = 0.0f;
    #pragma unroll
    for (int k2 = 0; k2 < TK; ++k2) { v[k2] = __expf(v[k2] - mx); s += v[k2]; }
    float inv = 1.0f / s;
    #pragma unroll
    for (int k2 = 0; k2 < TK; ++k2) w_out[tid * TK + k2] = v[k2] * inv;
  }
}

// -------------------------- stage 4: weighted circular gather --------------
// out[b,l,h,e] = sum_k w[b,k] * values[b, (l - idx[k]) mod L, h, e]
__global__ __launch_bounds__(256)
void k_agg(const float* __restrict__ V, const int* __restrict__ idx,
           const float* __restrict__ w, float* __restrict__ out) {
  __shared__ int   s_idx[TK];
  __shared__ float s_w[NB][TK];
  const int tid = threadIdx.x;
  if (tid < TK) s_idx[tid] = idx[tid];
  if (tid < NB * TK) s_w[tid / TK][tid % TK] = w[tid];
  __syncthreads();

  const unsigned gid = blockIdx.x * 256u + tid;   // over B*L*H*(E/4) = 2^21
  const int e4 = gid & 15;
  const int h  = (gid >> 4) & 7;
  const int l  = (gid >> 7) & (SEQ - 1);
  const int b  = gid >> 18;

  const float4* Vb = (const float4*)V;
  float4 acc = {0.0f, 0.0f, 0.0f, 0.0f};
  #pragma unroll
  for (int k2 = 0; k2 < TK; ++k2) {
    const int sl = (l + SEQ - s_idx[k2]) & (SEQ - 1);
    const float4 vv = Vb[((((size_t)b * SEQ + sl) * 8 + h) << 4) + e4];
    const float  ww = s_w[b][k2];
    acc.x += ww * vv.x; acc.y += ww * vv.y; acc.z += ww * vv.z; acc.w += ww * vv.w;
  }
  ((float4*)out)[((((size_t)b * SEQ + l) * 8 + h) << 4) + e4] = acc;
}

// ---------------------------------------------------------------------------
extern "C" void kernel_launch(void* const* d_in, const int* in_sizes, int n_in,
                              void* d_out, int out_size, void* d_ws, size_t ws_size,
                              hipStream_t stream) {
  const float* q = (const float*)d_in[0];
  const float* k = (const float*)d_in[1];
  const float* v = (const float*)d_in[2];
  // d_in[3] = attn_mask (unused by reference path)

  const size_t elems  = (size_t)NB * SEQ * CHN;       // 8,388,608
  const size_t pbytes = (size_t)NB * NT * 31 * sizeof(float);
  char* ws = (char*)d_ws;

  const bool use_bf16_ws = ws_size >= elems * 4 + pbytes + 256;

  float* P;
  int*   idx;
  float* w;
  if (use_bf16_ws) {
    bf16_t* qbf = (bf16_t*)ws;
    bf16_t* kbf = (bf16_t*)(ws + elems * 2);
    P   = (float*)(ws + elems * 4);
    idx = (int*)(ws + elems * 4 + pbytes);
    w   = (float*)((char*)idx + 64);

    const int n8 = (int)(elems / 8);                   // 1,048,576
    k_cvt_bf16<<<dim3(n8 / 256), 256, 0, stream>>>(q, qbf, n8);
    k_cvt_bf16<<<dim3(n8 / 256), 256, 0, stream>>>(k, kbf, n8);
    k_corr_bf16<<<dim3(NT / 4, NB), 256, 0, stream>>>(qbf, kbf, P);
  } else {
    P   = (float*)ws;
    idx = (int*)(ws + pbytes);
    w   = (float*)((char*)idx + 64);
    k_corr_f32<<<dim3(NT, NB), 128, 0, stream>>>(q, k, P);
  }

  k_topk<<<1, 256, 0, stream>>>(P, idx, w);

  const int total = NB * SEQ * 8 * 16;                 // 2,097,152 threads
  k_agg<<<total / 256, 256, 0, stream>>>(v, idx, w, (float*)d_out);
}
